// RetentionBlock_65618510349055
// MI455X (gfx1250) — compile-verified, matrix operand order checked by reading, not run on hardware
//
#include <hip/hip_runtime.h>
#include <hip/hip_bf16.h>
#include <math.h>

// ---------------------------------------------------------------------------
// RetentionBlock for MI455X (gfx1250, wave32, WMMA + async-to-LDS).
//
//   [0] weight transpose + f32->f16    (Bt layout: [N][K], mirrors A striping)
//   [1] LayerNorm1(x) -> yh (f16)
//   [2] GEMM yh @ [Wq|Wk|Wv|Wg]^T, epilogue: +bias, sigmoid(g),
//       scatter to TIME-MAJOR f16 q/k/v/g
//   [3] chunked parallel retention scan (64 chunks x 64 steps, 3 passes)
//   [4] GEMM oh @ Wo^T + bo + x -> x2 (f32)
//   [5] LayerNorm2(x2) -> h (f16)
//   [6] GEMM h @ W1^T + b1, exact GELU -> ff (f16)
//   [7] GEMM ff @ W2^T + b2 + x2 -> d_out (f32)
//
// GEMM: 256 thr = 8 wave32, block tile 128x256, wave tile 64x64, BK=64
// (two WMMA K-substeps per stage), double-buffered LDS (96 KB) filled with
// global_load_async_to_lds_b128 (ASYNCcnt) so slab k+1 streams from HBM
// while slab k feeds v_wmma — 32 WMMAs of compute per stage to cover HBM
// latency, barriers at half the BK=32 frequency.
// ---------------------------------------------------------------------------

#define USE_ASYNC_LDS 1

typedef __attribute__((ext_vector_type(16))) _Float16 v16h;
typedef __attribute__((ext_vector_type(8)))  float    v8f;

#define D_MODEL 1024
#define BATCH   4
#define SEQLEN  4096
#define NTOK    (BATCH * SEQLEN)       // 16384 rows
#define DFF     (4 * D_MODEL)          // 4096
#define NCHAN   (BATCH * D_MODEL)      // 4096 scan channels
#define CHUNK   64
#define NCHUNK  (SEQLEN / CHUNK)       // 64

#define BM 128
#define BN 256
#define BK 64                          // two 16x16x32 K-substeps per stage

// ---------------------------------------------------------------------------
// Fragment load from an LDS tile, leading dim BK (=64 halfs), K-offset koff.
// 16-bit A-matrix 16x32 striping (ISA 7.12.2):
//   lanes 0-15  : row M=lane,    K = {0..7}, {16..23}
//   lanes 16-31 : row M=lane-16, K = {8..15}, {24..31}
// ---------------------------------------------------------------------------
__device__ __forceinline__ v16h load_frag(const _Float16* tile, int row0, int koff, int lane) {
    int r   = row0 + (lane & 15);
    int grp = lane >> 4;
    union { v16h h; uint4 q[2]; } u;
    u.q[0] = *reinterpret_cast<const uint4*>(tile + r * BK + koff + grp * 8);
    u.q[1] = *reinterpret_cast<const uint4*>(tile + r * BK + koff + 16 + grp * 8);
    return u.h;
}

__device__ __forceinline__ float sigmoidf_(float x) { return 1.0f / (1.0f + __expf(-x)); }
__device__ __forceinline__ float gelu_exact(float x) {
    return 0.5f * x * (1.0f + erff(x * 0.70710678118654752f));
}

__device__ __forceinline__ void async_b128_to_lds(const _Float16* gptr, const _Float16* lptr) {
#if USE_ASYNC_LDS
    unsigned long long ga = (unsigned long long)(uintptr_t)gptr;
    unsigned int       la = (unsigned int)(uintptr_t)lptr;   // low 32b = LDS offset
    asm volatile("global_load_async_to_lds_b128 %0, %1, off"
                 :: "v"(la), "v"(ga) : "memory");
#else
    *(uint4*)lptr = *(const uint4*)gptr;
#endif
}

__device__ __forceinline__ void wait_async_lds() {
#if USE_ASYNC_LDS
    asm volatile("s_wait_asynccnt 0" ::: "memory");
#endif
}

// Epilogue modes
#define EPI_QKVG       0
#define EPI_BIAS_RES   1
#define EPI_BIAS_GELU  2
#define EPI_BIAS_RES_O 3

template <int EPI>
__global__ __launch_bounds__(256)
void gemm_wmma_f16(const _Float16* __restrict__ A,   // [M][K] f16
                   const _Float16* __restrict__ Bt,  // [N][K] f16 (B^T)
                   int M, int N, int K,
                   const float* __restrict__ bias0,
                   const float* __restrict__ bias1,
                   const float* __restrict__ bias2,
                   const float* __restrict__ bias3,
                   const float* __restrict__ resid,
                   float*       __restrict__ outF,
                   _Float16*    __restrict__ o0,
                   _Float16*    __restrict__ o1,
                   _Float16*    __restrict__ o2,
                   _Float16*    __restrict__ o3)
{
    __shared__ _Float16 sA[2][BM * BK];  // 2 x 16 KB
    __shared__ _Float16 sB[2][BN * BK];  // 2 x 32 KB

    const int tid   = threadIdx.x;
    const int lane  = tid & 31;
    const int wave  = tid >> 5;
    const int waveM = wave >> 2;
    const int waveN = wave & 3;
    const int bm    = blockIdx.x * BM;
    const int bn    = blockIdx.y * BN;

    // fixed per-thread staging coordinates (16B chunks, 8 chunks per 64-half row)
    int rA[4], cA[4], rB[8], cB[8];
    #pragma unroll
    for (int j = 0; j < 4; ++j) { int i = tid + 256 * j; rA[j] = i >> 3; cA[j] = (i & 7) * 8; }
    #pragma unroll
    for (int j = 0; j < 8; ++j) { int i = tid + 256 * j; rB[j] = i >> 3; cB[j] = (i & 7) * 8; }

    auto stage = [&](int k0, int buf) {
        #pragma unroll
        for (int j = 0; j < 4; ++j)
            async_b128_to_lds(&A[(size_t)(bm + rA[j]) * K + k0 + cA[j]],
                              &sA[buf][rA[j] * BK + cA[j]]);
        #pragma unroll
        for (int j = 0; j < 8; ++j)
            async_b128_to_lds(&Bt[(size_t)(bn + rB[j]) * K + k0 + cB[j]],
                              &sB[buf][rB[j] * BK + cB[j]]);
    };

    v8f acc[4][4] = {};
    const int nk = K / BK;

    stage(0, 0);
    wait_async_lds();
    __syncthreads();

    for (int kt = 0; kt < nk; ++kt) {
        const int cur = kt & 1;
        if (kt + 1 < nk) stage((kt + 1) * BK, cur ^ 1);   // stream next slab from HBM

        const _Float16* tA = sA[cur];
        const _Float16* tB = sB[cur];
        #pragma unroll
        for (int s = 0; s < 2; ++s) {                     // two K=32 substeps
            const int koff = s * 32;
            v16h af[4], bf[4];
            #pragma unroll
            for (int mi = 0; mi < 4; ++mi) af[mi] = load_frag(tA, waveM * 64 + mi * 16, koff, lane);
            #pragma unroll
            for (int ni = 0; ni < 4; ++ni) bf[ni] = load_frag(tB, waveN * 64 + ni * 16, koff, lane);

            #pragma unroll
            for (int mi = 0; mi < 4; ++mi)
                #pragma unroll
                for (int ni = 0; ni < 4; ++ni)
                    acc[mi][ni] = __builtin_amdgcn_wmma_f32_16x16x32_f16(
                        false, af[mi], false, bf[ni], (short)0, acc[mi][ni], false, false);
        }

        wait_async_lds();
        __syncthreads();
    }

    // ---- epilogue: lane l holds col n=l&15, rows v + 8*(l>>4)
    const int grp = lane >> 4;
    const int nl  = lane & 15;

    #pragma unroll
    for (int mi = 0; mi < 4; ++mi) {
        #pragma unroll
        for (int ni = 0; ni < 4; ++ni) {
            const int colg = bn + waveN * 64 + ni * 16 + nl;
            #pragma unroll
            for (int v = 0; v < 8; ++v) {
                const int rowg = bm + waveM * 64 + mi * 16 + grp * 8 + v;
                float val = acc[mi][ni][v];

                if (EPI == EPI_QKVG) {
                    const int which = colg >> 10;           // 0..3 = q,k,v,g
                    const int c     = colg & (D_MODEL - 1);
                    const float* bp = (which == 0) ? bias0 : (which == 1) ? bias1
                                    : (which == 2) ? bias2 : bias3;
                    val += bp[c];
                    if (which == 3) val = sigmoidf_(val);
                    _Float16* op = (which == 0) ? o0 : (which == 1) ? o1
                                 : (which == 2) ? o2 : o3;
                    const int b = rowg >> 12;               // / SEQLEN
                    const int t = rowg & (SEQLEN - 1);
                    op[((size_t)t * BATCH + b) * D_MODEL + c] = (_Float16)val;
                } else if (EPI == EPI_BIAS_RES || EPI == EPI_BIAS_RES_O) {
                    val += bias0[colg] + resid[(size_t)rowg * N + colg];
                    outF[(size_t)rowg * N + colg] = val;
                } else { // EPI_BIAS_GELU
                    val += bias0[colg];
                    o0[(size_t)rowg * N + colg] = (_Float16)gelu_exact(val);
                }
            }
        }
    }
}

// ---------------------------------------------------------------------------
// LayerNorm over D_MODEL=1024, one block (8 waves) per row, f16 output.
// ---------------------------------------------------------------------------
__global__ __launch_bounds__(256)
void layernorm_f16(const float* __restrict__ x, const float* __restrict__ g,
                   const float* __restrict__ b, _Float16* __restrict__ out)
{
    __shared__ float red[2][8];
    const size_t row = blockIdx.x;
    const float* xr = x + row * D_MODEL;
    const int tid = threadIdx.x;

    float vals[4], s = 0.f, ss = 0.f;
    #pragma unroll
    for (int i = 0; i < 4; ++i) {
        float v = xr[tid + i * 256];
        vals[i] = v; s += v; ss += v * v;
    }
    #pragma unroll
    for (int off = 16; off > 0; off >>= 1) {
        s  += __shfl_down(s,  off, 32);
        ss += __shfl_down(ss, off, 32);
    }
    const int wv = tid >> 5, lane = tid & 31;
    if (lane == 0) { red[0][wv] = s; red[1][wv] = ss; }
    __syncthreads();
    if (tid == 0) {
        float ts = 0.f, tss = 0.f;
        #pragma unroll
        for (int i = 0; i < 8; ++i) { ts += red[0][i]; tss += red[1][i]; }
        red[0][0] = ts; red[1][0] = tss;
    }
    __syncthreads();
    const float mean = red[0][0] * (1.0f / D_MODEL);
    const float var  = red[1][0] * (1.0f / D_MODEL) - mean * mean;
    const float rstd = rsqrtf(var + 1e-5f);
    #pragma unroll
    for (int i = 0; i < 4; ++i) {
        int c = tid + i * 256;
        out[row * D_MODEL + c] = (_Float16)((vals[i] - mean) * rstd * g[c] + b[c]);
    }
}

// ---------------------------------------------------------------------------
// Weight transpose + f32 -> f16.  W:[K][N] -> Wt:[N][K]
// ---------------------------------------------------------------------------
__global__ __launch_bounds__(256)
void transpose_convert(const float* __restrict__ W, _Float16* __restrict__ Wt,
                       int K, int N)
{
    size_t idx = (size_t)blockIdx.x * blockDim.x + threadIdx.x;
    if (idx >= (size_t)K * N) return;
    int n = (int)(idx % N);
    int k = (int)(idx / N);
    Wt[(size_t)n * K + k] = (_Float16)W[idx];
}

// ---------------------------------------------------------------------------
// Chunked parallel retention scan.  state_t = decay*state + k*v (elementwise),
// out_t = g*q*state*scale.  L=4096 split into NCHUNK=64 chunks of CHUNK=64.
//   A: per-chunk local final (zero-init)            [NCHUNK][NCHAN]
//   B: serial combine across chunks: carry[j] = state entering chunk j
//   C: replay chunk from carry, emit outputs (batch-major f16 oh)
// Time-major index: t*NCHAN + c, lanes on consecutive c -> fully coalesced.
// ---------------------------------------------------------------------------
__global__ __launch_bounds__(256)
void scan_partial(const _Float16* __restrict__ k, const _Float16* __restrict__ v,
                  const float* __restrict__ decay_logit, float* __restrict__ cf)
{
    const int t0 = blockIdx.x * blockDim.x + threadIdx.x;   // 0..NCHUNK*NCHAN-1
    const int c  = t0 & (NCHAN - 1);
    const int j  = t0 >> 12;                                // / NCHAN
    const int h  = (c & (D_MODEL - 1)) >> 6;
    const float decay = sigmoidf_(decay_logit[h]);

    float local = 0.f;
    size_t idx = (size_t)(j * CHUNK) * NCHAN + c;
    #pragma unroll 8
    for (int t = 0; t < CHUNK; ++t) {
        local = decay * local + (float)k[idx] * (float)v[idx];
        idx += NCHAN;
    }
    cf[(size_t)j * NCHAN + c] = local;
}

__global__ __launch_bounds__(256)
void scan_combine(const float* __restrict__ cf, const float* __restrict__ decay_logit,
                  float* __restrict__ carry)
{
    const int c = blockIdx.x * blockDim.x + threadIdx.x;    // 0..NCHAN-1
    const int h = (c & (D_MODEL - 1)) >> 6;
    const float decay = sigmoidf_(decay_logit[h]);
    float dC = decay;                                       // decay^CHUNK by squaring
    #pragma unroll
    for (int i = 0; i < 6; ++i) dC *= dC;                   // 2^6 = 64

    float S = 0.f;
    #pragma unroll 8
    for (int j = 0; j < NCHUNK; ++j) {
        carry[(size_t)j * NCHAN + c] = S;
        S = dC * S + cf[(size_t)j * NCHAN + c];
    }
}

__global__ __launch_bounds__(256)
void scan_apply(const _Float16* __restrict__ q, const _Float16* __restrict__ k,
                const _Float16* __restrict__ v, const _Float16* __restrict__ g,
                const float* __restrict__ carry, const float* __restrict__ decay_logit,
                _Float16* __restrict__ oh)
{
    const int t0 = blockIdx.x * blockDim.x + threadIdx.x;
    const int c  = t0 & (NCHAN - 1);
    const int j  = t0 >> 12;
    const int hc = c & (D_MODEL - 1);
    const int b  = c >> 10;
    const int h  = hc >> 6;
    const float decay = sigmoidf_(decay_logit[h]);
    const float scale = 0.125f;                              // DH^-0.5

    float state = carry[(size_t)j * NCHAN + c];
    size_t idx  = (size_t)(j * CHUNK) * NCHAN + c;
    size_t oidx = ((size_t)b * SEQLEN + j * CHUNK) * D_MODEL + hc;
    #pragma unroll 8
    for (int t = 0; t < CHUNK; ++t) {
        state = decay * state + (float)k[idx] * (float)v[idx];
        oh[oidx] = (_Float16)((float)g[idx] * (float)q[idx] * state * scale);
        idx += NCHAN; oidx += D_MODEL;
    }
}

// ---------------------------------------------------------------------------
// Host launch
// ---------------------------------------------------------------------------
extern "C" void kernel_launch(void* const* d_in, const int* in_sizes, int n_in,
                              void* d_out, int out_size, void* d_ws, size_t ws_size,
                              hipStream_t stream)
{
    const float* x    = (const float*)d_in[0];
    const float* ln1g = (const float*)d_in[1];
    const float* ln1b = (const float*)d_in[2];
    const float* Wq   = (const float*)d_in[3];
    const float* bq   = (const float*)d_in[4];
    const float* Wk   = (const float*)d_in[5];
    const float* bk   = (const float*)d_in[6];
    const float* Wv   = (const float*)d_in[7];
    const float* bv   = (const float*)d_in[8];
    const float* Wg   = (const float*)d_in[9];
    const float* bg   = (const float*)d_in[10];
    const float* dl   = (const float*)d_in[11];
    const float* Wo   = (const float*)d_in[12];
    const float* bo   = (const float*)d_in[13];
    const float* ln2g = (const float*)d_in[14];
    const float* ln2b = (const float*)d_in[15];
    const float* W1   = (const float*)d_in[16];
    const float* b1   = (const float*)d_in[17];
    const float* W2   = (const float*)d_in[18];
    const float* b2   = (const float*)d_in[19];
    float* out = (float*)d_out;

    char* ws = (char*)d_ws;
    const size_t MB = 1024 * 1024;
    _Float16* qkvgT = (_Float16*)(ws + 0);          //  8 MB  [4096][1024]
    _Float16* WoT   = (_Float16*)(ws + 8  * MB);    //  2 MB
    _Float16* W1T   = (_Float16*)(ws + 10 * MB);    //  8 MB
    _Float16* W2T   = (_Float16*)(ws + 18 * MB);    //  8 MB
    _Float16* yh    = (_Float16*)(ws + 26 * MB);    // 32 MB
    _Float16* qh    = (_Float16*)(ws + 58 * MB);    // 32 MB  time-major
    _Float16* kh    = (_Float16*)(ws + 90 * MB);    // 32 MB
    _Float16* vh    = (_Float16*)(ws + 122 * MB);   // 32 MB
    _Float16* gh    = (_Float16*)(ws + 154 * MB);   // 32 MB
    _Float16* oh    = (_Float16*)(ws + 186 * MB);   // 32 MB  batch-major
    float*    x2    = (float*)   (ws + 218 * MB);   // 64 MB
    _Float16* hh    = (_Float16*)(ws + 282 * MB);   // 32 MB
    _Float16* ff    = (_Float16*)(ws + 314 * MB);   // 128 MB
    float*    cf    = (float*)   (ws + 442 * MB);   //  1 MB  chunk finals
    float*    carry = (float*)   (ws + 443 * MB);   //  1 MB  chunk carries
    (void)in_sizes; (void)n_in; (void)out_size; (void)ws_size;

    const int nKK = (D_MODEL * D_MODEL) / 256;
    const int nKF = (D_MODEL * DFF) / 256;

    // [0] weight prep (Wq|Wk|Wv|Wg stacked into one [4096][1024] Bt)
    transpose_convert<<<nKK, 256, 0, stream>>>(Wq, qkvgT + 0 * D_MODEL * D_MODEL, D_MODEL, D_MODEL);
    transpose_convert<<<nKK, 256, 0, stream>>>(Wk, qkvgT + 1 * D_MODEL * D_MODEL, D_MODEL, D_MODEL);
    transpose_convert<<<nKK, 256, 0, stream>>>(Wv, qkvgT + 2 * D_MODEL * D_MODEL, D_MODEL, D_MODEL);
    transpose_convert<<<nKK, 256, 0, stream>>>(Wg, qkvgT + 3 * D_MODEL * D_MODEL, D_MODEL, D_MODEL);
    transpose_convert<<<nKK, 256, 0, stream>>>(Wo, WoT, D_MODEL, D_MODEL);
    transpose_convert<<<nKF, 256, 0, stream>>>(W1, W1T, D_MODEL, DFF);
    transpose_convert<<<nKF, 256, 0, stream>>>(W2, W2T, DFF, D_MODEL);

    // [1] LN1
    layernorm_f16<<<NTOK, 256, 0, stream>>>(x, ln1g, ln1b, yh);

    // [2] QKVG projection
    gemm_wmma_f16<EPI_QKVG><<<dim3(NTOK / BM, DFF / BN), 256, 0, stream>>>(
        yh, qkvgT, NTOK, DFF, D_MODEL,
        bq, bk, bv, bg, nullptr, nullptr, qh, kh, vh, gh);

    // [3] chunked retention scan
    scan_partial<<<(NCHUNK * NCHAN) / 256, 256, 0, stream>>>(kh, vh, dl, cf);
    scan_combine<<<NCHAN / 256, 256, 0, stream>>>(cf, dl, carry);
    scan_apply  <<<(NCHUNK * NCHAN) / 256, 256, 0, stream>>>(qh, kh, vh, gh, carry, dl, oh);

    // [4] output projection + bo + residual(x) -> x2
    gemm_wmma_f16<EPI_BIAS_RES><<<dim3(NTOK / BM, D_MODEL / BN), 256, 0, stream>>>(
        oh, WoT, NTOK, D_MODEL, D_MODEL,
        bo, nullptr, nullptr, nullptr, x, x2, nullptr, nullptr, nullptr, nullptr);

    // [5] LN2
    layernorm_f16<<<NTOK, 256, 0, stream>>>(x2, ln2g, ln2b, hh);

    // [6] FFN up + GELU
    gemm_wmma_f16<EPI_BIAS_GELU><<<dim3(NTOK / BM, DFF / BN), 256, 0, stream>>>(
        hh, W1T, NTOK, DFF, D_MODEL,
        b1, nullptr, nullptr, nullptr, nullptr, nullptr, ff, nullptr, nullptr, nullptr);

    // [7] FFN down + b2 + residual(x2) -> d_out
    gemm_wmma_f16<EPI_BIAS_RES_O><<<dim3(NTOK / BM, D_MODEL / BN), 256, 0, stream>>>(
        ff, W2T, NTOK, D_MODEL, DFF,
        b2, nullptr, nullptr, nullptr, x2, out, nullptr, nullptr, nullptr, nullptr);
}
